// Net_23587960389992
// MI455X (gfx1250) — matrix operand order repaired and checked by reading, not compiled, not run on hardware
//
#include <hip/hip_runtime.h>
#include <hip/hip_bf16.h>
#include <cmath>

// ---------------- problem constants ----------------
#define NN      100000
#define EE      3200000
#define F_IN    512
#define HH      256
#define CC      64
#define LL      8
#define ALPHA   0.1f
#define THETA   0.5f

// ---------------- WMMA types ----------------
typedef __attribute__((ext_vector_type(16))) __bf16 v16bf;
typedef __attribute__((ext_vector_type(8)))  float  v8f;
static_assert(sizeof(v16bf) == 32, "v16bf must be 32B");

// GEMM tiling
#define BM 64
#define BN 256
#define A_STRIDE 80   // bytes per LDS A row (32 bf16 = 64B + 16B pad, keeps 16B align)
#define B_STRIDE 80   // bytes per LDS B column-run (32 bf16 + pad)

// f32 -> bf16 (round to nearest even)
__device__ __forceinline__ unsigned short f2bf(float f) {
    unsigned u = __float_as_uint(f);
    u = (u + 0x7FFFu + ((u >> 16) & 1u)) >> 16;
    return (unsigned short)u;
}
__device__ __forceinline__ unsigned pk_bf16(float a, float b) {
#if __has_builtin(__builtin_amdgcn_cvt_pk_bf16_f32)
    typedef __attribute__((ext_vector_type(2))) __bf16 v2bf;
    union { v2bf v; unsigned u; } cv;
    cv.v = __builtin_amdgcn_cvt_pk_bf16_f32(a, b);   // D[15:0]=a, D[31:16]=b
    return cv.u;
#else
    return (unsigned)f2bf(a) | ((unsigned)f2bf(b) << 16);
#endif
}

// Async global->LDS 16B copy (GLOBAL_LOAD_ASYNC_TO_LDS_B128, tracked by ASYNCcnt)
__device__ __forceinline__ void async_g2l_b128(unsigned ldsOff, const void* gaddr) {
    asm volatile("global_load_async_to_lds_b128 %0, %1, off"
                 :: "v"(ldsOff), "v"(gaddr) : "memory");
}
__device__ __forceinline__ void wait_async0() {
    asm volatile("s_wait_asynccnt 0" ::: "memory");
}

// ---------------------------------------------------------------------------
// Weight cast + swizzle into B-fragment-friendly layout:
//   dst[(kb*256 + n)*32 + kk] = bf16( W[(kb*32+kk)*256 + n] )
// Region 0: W_in (K=512 -> kb 0..15), Region 1: conv_W (8 layers, kb 0..7 each)
// ---------------------------------------------------------------------------
__global__ void cast_w_kernel(const float* __restrict__ Win,
                              const float* __restrict__ convW,
                              unsigned short* __restrict__ dst) {
    int idx = blockIdx.x * blockDim.x + threadIdx.x;     // 0 .. 655359
    if (idx >= (F_IN * HH + LL * HH * HH)) return;
    float v;
    if (idx < F_IN * HH) {                               // W_in region
        int kk = idx & 31, n = (idx >> 5) & 255, kb = idx >> 13;
        v = Win[(kb * 32 + kk) * HH + n];
    } else {
        int j = idx - F_IN * HH;
        int l = j >> 16, r = j & 65535;
        int kk = r & 31, n = (r >> 5) & 255, kb = r >> 13;
        v = convW[(size_t)l * HH * HH + (kb * 32 + kk) * HH + n];
    }
    dst[idx] = f2bf(v);
}

// ---------------------------------------------------------------------------
// Fused WMMA GEMM:  out = relu( ca * (A @ W) + cb * Csrc + bias )
//   A    : [nrows x K] f32 (converted to bf16 on the fly)
//   Wswz : swizzled bf16 weights [K/32][256][32]
//   Workgroup = 256 threads (8 waves) computes BM=64 rows x 256 cols.
//   B block staged with async global->LDS; A block converted through VALU.
// ---------------------------------------------------------------------------
__launch_bounds__(256)
__global__ void gemm_bf16_wmma(const float* __restrict__ A,
                               const unsigned short* __restrict__ Wswz,
                               const float* __restrict__ bias,   // [256] or null
                               const float* __restrict__ Csrc,   // [nrows x 256] or null
                               float* __restrict__ out,
                               float* __restrict__ out2,         // optional duplicate
                               int nrows, int K, float ca, float cb) {
    __shared__ unsigned char lds[BM * A_STRIDE + BN * B_STRIDE]; // 5120 + 20480 = 25600 B
    unsigned char* ldsA = lds;
    unsigned char* ldsB = lds + BM * A_STRIDE;

    const int tid  = threadIdx.x;
    const int wave = tid >> 5;
    const int lane = tid & 31;
    const int laneLo = lane & 15;
    const int laneHi = lane >> 4;           // 0 | 1
    const int rt = wave & 3;                // row tile 0..3 (16 rows each)
    const int ch = wave >> 2;               // column half 0..1 (128 cols each)
    const int rowBlock = blockIdx.x * BM;

    const unsigned ldsBOffBase = (unsigned)(size_t)(ldsB + tid * B_STRIDE);

    v8f acc[8] = {};                        // 8 C-tiles of 16x16 f32

    const int nKB = K >> 5;
    for (int kb = 0; kb < nKB; ++kb) {
        __syncthreads();                    // protect LDS reuse
        // ---- async-stage B block: 256 columns x 32 K bf16 (pre-swizzled) ----
        {
            const unsigned char* g = (const unsigned char*)Wswz +
                                     ((size_t)kb * BN * 32) * 2 + (size_t)tid * 64;
            #pragma unroll
            for (int i = 0; i < 4; ++i)
                async_g2l_b128(ldsBOffBase + i * 16, g + i * 16);
        }
        // ---- stage A block [64 rows x 32 K] f32 -> bf16 LDS (VALU convert) ----
        #pragma unroll
        for (int i = 0; i < 2; ++i) {
            int idx = tid + i * 256;        // float4 index over [64][8]
            int r  = idx >> 3;
            int c4 = idx & 7;
            int grow = rowBlock + r;
            float4 v = make_float4(0.f, 0.f, 0.f, 0.f);
            if (grow < nrows)
                v = *(const float4*)(A + (size_t)grow * K + kb * 32 + c4 * 4);
            *(uint2*)(ldsA + r * A_STRIDE + c4 * 8) =
                make_uint2(pk_bf16(v.x, v.y), pk_bf16(v.z, v.w));
        }
        wait_async0();                      // our async B copies landed in LDS
        __syncthreads();

        // ---- A fragment (16x32 bf16, per ISA layout) ----
        union { uint4 q[2]; v16bf v; } afr;
        const unsigned char* abase = ldsA + (rt * 16 + laneLo) * A_STRIDE;
        afr.q[0] = *(const uint4*)(abase + laneHi * 16);        // K 0-7 | 8-15
        afr.q[1] = *(const uint4*)(abase + 32 + laneHi * 16);   // K 16-23 | 24-31

        // ---- 8 column tiles: B frag + WMMA ----
        #pragma unroll
        for (int t = 0; t < 8; ++t) {
            int n = ch * 128 + t * 16 + laneLo;
            union { uint4 q[2]; v16bf v; } bfr;
            const uint4* bbase = (const uint4*)(ldsB + n * B_STRIDE + laneHi * 16);
            bfr.q[0] = bbase[0];
            bfr.q[1] = bbase[2];
            acc[t] = __builtin_amdgcn_wmma_f32_16x16x32_bf16(
                         false, afr.v, false, bfr.v,
                         (short)0, acc[t], false, false);
        }
    }

    // ---- epilogue: combine, ReLU, store ----
    #pragma unroll
    for (int t = 0; t < 8; ++t) {
        int col = ch * 128 + t * 16 + laneLo;
        #pragma unroll
        for (int v = 0; v < 8; ++v) {
            int row = rowBlock + rt * 16 + laneHi * 8 + v;
            if (row < nrows) {
                float r = ca * acc[t][v];
                if (Csrc) r += cb * Csrc[(size_t)row * BN + col];
                if (bias) r += bias[col];
                r = fmaxf(r, 0.f);
                out[(size_t)row * BN + col] = r;
                if (out2) out2[(size_t)row * BN + col] = r;
            }
        }
    }
}

// ---------------------------------------------------------------------------
// SpMM: agg[row] += val * h[col]   (one wave per edge, 8 floats per lane)
// h (100 MB) fits in the 192 MB L2 -> gathers & atomics served from L2.
// ---------------------------------------------------------------------------
__global__ void spmm_kernel(const int* __restrict__ rows,
                            const int* __restrict__ cols,
                            const float* __restrict__ vals,
                            const float* __restrict__ feat,
                            float* __restrict__ agg, int nedges) {
    int e = blockIdx.x * 8 + (threadIdx.x >> 5);
    if (e >= nedges) return;
    int lane = threadIdx.x & 31;
    int r = rows[e], c = cols[e];
    float v = vals[e];
    const float* src = feat + (size_t)c * HH + lane * 8;
    float*       dst = agg  + (size_t)r * HH + lane * 8;
    float4 a = *(const float4*)(src);
    float4 b = *(const float4*)(src + 4);
    unsafeAtomicAdd(dst + 0, v * a.x);
    unsafeAtomicAdd(dst + 1, v * a.y);
    unsafeAtomicAdd(dst + 2, v * a.z);
    unsafeAtomicAdd(dst + 3, v * a.w);
    unsafeAtomicAdd(dst + 4, v * b.x);
    unsafeAtomicAdd(dst + 5, v * b.y);
    unsafeAtomicAdd(dst + 6, v * b.z);
    unsafeAtomicAdd(dst + 7, v * b.w);
}

// mix = (1-ALPHA)*agg + ALPHA*x0   (in place into agg)
__global__ void mix_kernel(float* __restrict__ agg, const float* __restrict__ x0,
                           long nfloats) {
    long i = ((long)blockIdx.x * blockDim.x + threadIdx.x) * 4;
    if (i >= nfloats) return;
    float4 a = *(float4*)(agg + i);
    float4 b = *(const float4*)(x0 + i);
    a.x = (1.f - ALPHA) * a.x + ALPHA * b.x;
    a.y = (1.f - ALPHA) * a.y + ALPHA * b.y;
    a.z = (1.f - ALPHA) * a.z + ALPHA * b.z;
    a.w = (1.f - ALPHA) * a.w + ALPHA * b.w;
    *(float4*)(agg + i) = a;
}

// ---------------------------------------------------------------------------
// Output: logits = h @ W_out + b_out, then log_softmax over C=64.
// One wave per node; lane holds 2 classes; wave-shuffle reductions.
// ---------------------------------------------------------------------------
__global__ void out_kernel(const float* __restrict__ h,
                           const float* __restrict__ Wout,
                           const float* __restrict__ bout,
                           float* __restrict__ out, int nrows) {
    int node = blockIdx.x * 8 + (threadIdx.x >> 5);
    if (node >= nrows) return;
    int lane = threadIdx.x & 31;
    int c0 = lane * 2;
    float a0 = bout[c0], a1 = bout[c0 + 1];
    const float* hrow = h + (size_t)node * HH;
    for (int k = 0; k < HH; ++k) {
        float hv = hrow[k];
        float2 w = *(const float2*)(Wout + k * CC + c0);
        a0 = fmaf(hv, w.x, a0);
        a1 = fmaf(hv, w.y, a1);
    }
    float m = fmaxf(a0, a1);
    for (int o = 16; o > 0; o >>= 1) m = fmaxf(m, __shfl_xor(m, o, 32));
    float s = __expf(a0 - m) + __expf(a1 - m);
    for (int o = 16; o > 0; o >>= 1) s += __shfl_xor(s, o, 32);
    float lse = m + __logf(s);
    out[(size_t)node * CC + c0]     = a0 - lse;
    out[(size_t)node * CC + c0 + 1] = a1 - lse;
}

// ---------------------------------------------------------------------------
extern "C" void kernel_launch(void* const* d_in, const int* in_sizes, int n_in,
                              void* d_out, int out_size, void* d_ws, size_t ws_size,
                              hipStream_t stream) {
    const float* x        = (const float*)d_in[0];
    const float* edge_val = (const float*)d_in[1];
    const float* W_in     = (const float*)d_in[2];
    const float* b_in     = (const float*)d_in[3];
    const float* conv_W   = (const float*)d_in[4];
    const float* W_out    = (const float*)d_in[5];
    const float* b_out    = (const float*)d_in[6];
    const int*   edge_row = (const int*)d_in[7];
    const int*   edge_col = (const int*)d_in[8];

    char* ws = (char*)d_ws;
    const size_t featBytes = (size_t)NN * HH * sizeof(float);   // 102.4 MB
    float* x0  = (float*)(ws);
    float* h   = (float*)(ws + featBytes);
    float* agg = (float*)(ws + 2 * featBytes);
    unsigned short* Wswz = (unsigned short*)(ws + 3 * featBytes);
    const int nWElems = F_IN * HH + LL * HH * HH;               // 655360

    // 1) cast + swizzle all weights to bf16 B-fragment layout
    cast_w_kernel<<<(nWElems + 255) / 256, 256, 0, stream>>>(W_in, conv_W, Wswz);

    const int gemmBlocks = (NN + BM - 1) / BM;                  // 1563

    // 2) input GEMM: h = relu(x @ W_in + b_in); also write x0 copy
    gemm_bf16_wmma<<<gemmBlocks, 256, 0, stream>>>(
        x, Wswz, b_in, nullptr, h, x0, NN, F_IN, 1.f, 0.f);

    // 3) GCN2 layers
    const long nfeat = (long)NN * HH;
    for (int l = 0; l < LL; ++l) {
        hipMemsetAsync(agg, 0, featBytes, stream);
        spmm_kernel<<<(EE + 7) / 8, 256, 0, stream>>>(
            edge_row, edge_col, edge_val, h, agg, EE);
        mix_kernel<<<(int)((nfeat / 4 + 255) / 256), 256, 0, stream>>>(
            agg, x0, nfeat);
        float beta = logf(THETA / (float)(l + 1) + 1.0f);
        gemm_bf16_wmma<<<gemmBlocks, 256, 0, stream>>>(
            agg,                                  // A = mix
            Wswz + F_IN * HH + l * HH * HH,       // this layer's swizzled W
            nullptr,
            agg,                                  // Csrc = mix (identity mapping)
            h, nullptr, NN, HH, beta, 1.f - beta);
    }

    // 4) output GEMM + log_softmax
    out_kernel<<<(NN + 7) / 8, 256, 0, stream>>>(
        h, W_out, b_out, (float*)d_out, NN);
}
